// MultiRC_1864015807041
// MI455X (gfx1250) — matrix operand order repaired and checked by reading, not compiled, not run on hardware
//
#include <hip/hip_runtime.h>
#include <math.h>

typedef __attribute__((ext_vector_type(16))) _Float16 v16h;
typedef __attribute__((ext_vector_type(8)))  _Float16 v8h;
typedef __attribute__((ext_vector_type(8)))  float    v8f;

#define BB   64
#define TT   256
#define NIN  128
#define NHID 2048
// ALPHA = DECAY = THR = 0.5; ALPHA == 1-ALPHA lets us fuse r and U into one accumulator.
#define DECAY 0.5f
#define THR   0.5f

// ---------- fragment loaders (layouts per CDNA5 ISA §7.12.2) ----------

// A-matrix fragment, 16x32 f16, row-major source. `row` already points at (m, k0)
// for this lane's row m = lane&15; h = lane>>4 selects the K half-pattern.
__device__ __forceinline__ v16h load_a_frag(const _Float16* __restrict__ row, int h) {
  v8h lo = *(const v8h*)(row + h * 8);        // K = 8h .. 8h+7
  v8h hi = *(const v8h*)(row + 16 + h * 8);   // K = 16+8h .. 16+8h+7
  v16h a;
#pragma unroll
  for (int i = 0; i < 8; ++i) { a[i] = lo[i]; a[i + 8] = hi[i]; }
  return a;
}

// B-matrix fragment, 32x16 f16, from row-major [K x N] source with leading dim ld.
// lane l holds column N = nBase+(l&15); K = k0+8h+i (i<8), k0+16+8h+(i-8) (i>=8).
__device__ __forceinline__ v16h load_b_frag(const _Float16* __restrict__ src, int ld,
                                            int k0, int nBase, int lane) {
  const int hh = lane >> 4;
  const _Float16* p = src + (size_t)(k0 + 8 * hh) * ld + nBase + (lane & 15);
  v16h b;
#pragma unroll
  for (int i = 0; i < 8; ++i) { b[i] = p[(size_t)i * ld]; b[i + 8] = p[(size_t)(16 + i) * ld]; }
  return b;
}

// ---------- prep kernels ----------

__global__ void k_f32_to_f16(const float* __restrict__ src, _Float16* __restrict__ dst, int n) {
  int i = blockIdx.x * blockDim.x + threadIdx.x;
  if (i < n) dst[i] = (_Float16)src[i];
}

__global__ void k_init_state(const float* __restrict__ mem0, float* __restrict__ mem,
                             _Float16* __restrict__ s0, _Float16* __restrict__ s1, int n) {
  int i = blockIdx.x * blockDim.x + threadIdx.x;
  if (i < n) {
    mem[i] = mem0[i];
    s0[i]  = (_Float16)0.0f;
    s1[i]  = (_Float16)0.0f;
  }
}

// ---------- per-timestep fused kernel ----------
// Grid: (NHID/16, BB/16), block: 128 threads = 4 waves. The K reduction is
// split across the 4 waves (512 of spike@A each + 32 of x_t@W_in each), each
// via v_wmma_f32_16x16x32_f16; partials are reduced through LDS, then wave 0
// applies the pointwise SNN update for the 16x16 tile.
__global__ void __launch_bounds__(128)
snn_step(const _Float16* __restrict__ Ah,
         const _Float16* __restrict__ Wh,
         const _Float16* __restrict__ xh,
         const float*    __restrict__ bias,
         const _Float16* __restrict__ spike_cur,
         _Float16*       __restrict__ spike_next,
         float*          __restrict__ mem,
         float*          __restrict__ out_mems,
         float*          __restrict__ out_spikes,
         int t) {
  const int tid   = threadIdx.x;       // 0..127
  const int lane  = tid & 31;
  const int wave  = tid >> 5;          // 0..3
  const int nBase = blockIdx.x * 16;
  const int mBase = blockIdx.y * 16;
  const int mrow  = lane & 15;
  const int h     = lane >> 4;

  __shared__ float red[8][4][32];      // [acc elem][wave][lane], conflict-free

  v8f acc = {};

  // r = spike @ A : this wave covers K in [wave*512, wave*512+512)
  const _Float16* sprow = spike_cur + (size_t)(mBase + mrow) * NHID;
  const int kBeg = wave * (NHID / 4);
  const int kEnd = kBeg + (NHID / 4);
#pragma unroll 4
  for (int k0 = kBeg; k0 < kEnd; k0 += 32) {
    v16h a = load_a_frag(sprow + k0, h);
    v16h b = load_b_frag(Ah, NHID, k0, nBase, lane);
    acc = __builtin_amdgcn_wmma_f32_16x16x32_f16(false, a, false, b, (short)0, acc, false, false);
  }

  // U = x_t @ W_in : K = 128 split as one 32-slice per wave
  {
    const _Float16* xrow = xh + ((size_t)(mBase + mrow) * TT + t) * NIN;
    const int k0 = wave * 32;
    v16h a = load_a_frag(xrow + k0, h);
    v16h b = load_b_frag(Wh, NHID, k0, nBase, lane);
    acc = __builtin_amdgcn_wmma_f32_16x16x32_f16(false, a, false, b, (short)0, acc, false, false);
  }

  // Cross-wave reduction of the 4 partial accumulators through LDS.
#pragma unroll
  for (int r = 0; r < 8; ++r) red[r][wave][lane] = acc[r];
  __syncthreads();

  if (wave == 0) {
    // D layout: vgpr r, lane l -> n = nBase + (l&15), m = mBase + r + 8*(l>>4).
    const int   n  = nBase + (lane & 15);
    const float bn = bias[n];
#pragma unroll
    for (int r = 0; r < 8; ++r) {
      float sum = red[r][0][lane] + red[r][1][lane] + red[r][2][lane] + red[r][3][lane];
      const int    m   = mBase + r + 8 * h;
      const size_t idx = (size_t)m * NHID + n;
      float pre = 0.5f * (sum + bn);            // 0.5*r + 0.5*(U + bias)
      float y   = tanhf(pre);
      float sp  = (float)spike_cur[idx];
      float mo  = mem[idx];
      float mn  = mo * DECAY - THR * (1.0f - sp) + y;
      float sn  = (mn > THR) ? 1.0f : 0.0f;
      mem[idx]        = mn;
      spike_next[idx] = (_Float16)sn;
      const size_t oidx = ((size_t)m * TT + t) * NHID + n;
      out_mems[oidx]   = mn;
      out_spikes[oidx] = sn;
    }
  }
}

// ---------- host launcher ----------

extern "C" void kernel_launch(void* const* d_in, const int* in_sizes, int n_in,
                              void* d_out, int out_size, void* d_ws, size_t ws_size,
                              hipStream_t stream) {
  (void)in_sizes; (void)n_in; (void)out_size; (void)ws_size;

  const float* x    = (const float*)d_in[0];   // [64,256,128]
  const float* W_in = (const float*)d_in[1];   // [128,2048]
  const float* A    = (const float*)d_in[2];   // [2048,2048]
  const float* bias = (const float*)d_in[3];   // [2048]
  const float* mem0 = (const float*)d_in[4];   // [64,2048]

  // Workspace carve (all offsets 256B-aligned)
  char* ws = (char*)d_ws;
  _Float16* Ah  = (_Float16*)(ws);                              // 2048*2048*2 = 8388608
  _Float16* Wh  = (_Float16*)(ws + 8388608);                    // 128*2048*2  = 524288
  _Float16* xh  = (_Float16*)(ws + 8388608 + 524288);           // 64*256*128*2 = 4194304
  _Float16* sp0 = (_Float16*)(ws + 13107200);                   // 64*2048*2 = 262144
  _Float16* sp1 = (_Float16*)(ws + 13369344);                   // 64*2048*2 = 262144
  float*    mem = (float*)   (ws + 13631488);                   // 64*2048*4 = 524288

  float* out_mems   = (float*)d_out;
  float* out_spikes = out_mems + (size_t)BB * TT * NHID;

  const int nA = NHID * NHID;         // 4194304
  const int nW = NIN * NHID;          // 262144
  const int nX = BB * TT * NIN;       // 2097152
  const int nS = BB * NHID;           // 131072

  k_f32_to_f16<<<(nA + 255) / 256, 256, 0, stream>>>(A,    Ah, nA);
  k_f32_to_f16<<<(nW + 255) / 256, 256, 0, stream>>>(W_in, Wh, nW);
  k_f32_to_f16<<<(nX + 255) / 256, 256, 0, stream>>>(x,    xh, nX);
  k_init_state<<<(nS + 255) / 256, 256, 0, stream>>>(mem0, mem, sp0, sp1, nS);

  dim3 grid(NHID / 16, BB / 16);  // 128 x 4 tiles, 4 waves each -> 2048 waves/step
  dim3 block(128);
  for (int t = 0; t < TT; ++t) {
    _Float16* cur = (t & 1) ? sp1 : sp0;
    _Float16* nxt = (t & 1) ? sp0 : sp1;
    snn_step<<<grid, block, 0, stream>>>(Ah, Wh, xh, bias, cur, nxt, mem,
                                         out_mems, out_spikes, t);
  }
}
